// NormDenseVPL_40544491274895
// MI455X (gfx1250) — matrix operand order repaired and checked by reading, not compiled, not run on hardware
//
#include <hip/hip_runtime.h>
#include <hip/hip_bf16.h>
#include <stdint.h>

// Requires ~72MB workspace:
//  s_w2   : 4096 f32
//  s_inj2 : 4096 f32
//  A_hi/A_lo : 512*4096 bf16 each   (norm_x split)
//  Bt_hi/Bt_lo : 4096*4096 bf16 each (injected_pre^T split, [N][K] layout)

typedef __attribute__((ext_vector_type(16))) __bf16 v16bf;
typedef __attribute__((ext_vector_type(8)))  float  v8f;

#define EPSF 1e-5f

union FragBF { uint4 u[2]; v16bf v; };

#if defined(__AMDGCN__) && __has_builtin(__builtin_amdgcn_global_load_async_to_lds_b128)
#define USE_ASYNC_LDS 1
#else
#define USE_ASYNC_LDS 0
#endif

#if USE_ASYNC_LDS
// Builtin signature (from hipcc diagnostic): param0 = v4i in global AS, param1 = LDS, then imm offset, imm cpol.
typedef __attribute__((__vector_size__(4 * sizeof(int)))) int v4i;
typedef __attribute__((address_space(1))) v4i* gv4p;
typedef __attribute__((address_space(3))) v4i* lv4p;
// Copy 16B global -> LDS via the async DMA path (ASYNCcnt).
// lds_off is the 32-bit LDS byte offset (low 32 bits of the generic address,
// per the ISA's LDS aperture rule).
__device__ __forceinline__ void async_cp16(const void* g, unsigned lds_off) {
  __builtin_amdgcn_global_load_async_to_lds_b128(
      (gv4p)(uintptr_t)g, (lv4p)(uintptr_t)lds_off, 0, 0);
}
#if __has_builtin(__builtin_amdgcn_s_wait_asynccnt)
#define WAIT_ASYNC0() __builtin_amdgcn_s_wait_asynccnt(0)
#else
#define WAIT_ASYNC0() asm volatile("s_wait_asynccnt 0x0" ::: "memory")
#endif
#endif

__device__ __forceinline__ unsigned short f32_bf16_rne(float f) {
  unsigned u = __float_as_uint(f);
  unsigned r = u + 0x7fffu + ((u >> 16) & 1u);
  return (unsigned short)(r >> 16);
}
__device__ __forceinline__ float bf16_f32(unsigned short h) {
  return __uint_as_float(((unsigned)h) << 16);
}

// ---------------- K0: zero accumulators ----------------
__global__ void k_zero(float* __restrict__ p, int n) {
  int i = blockIdx.x * blockDim.x + threadIdx.x;
  if (i < n) p[i] = 0.f;
}

// ---------------- K1: row-normalize x, split to bf16 hi/lo ----------------
__global__ void k_norm_split_x(const float* __restrict__ x,
                               unsigned short* __restrict__ ah,
                               unsigned short* __restrict__ al) {
  __shared__ float red[256];
  const int row = blockIdx.x;
  const float* xr = x + (size_t)row * 4096;
  float s = 0.f;
  for (int i = threadIdx.x; i < 4096; i += 256) { float v = xr[i]; s += v * v; }
  red[threadIdx.x] = s;
  __syncthreads();
  for (int o = 128; o > 0; o >>= 1) {
    if (threadIdx.x < o) red[threadIdx.x] += red[threadIdx.x + o];
    __syncthreads();
  }
  const float inv = 1.f / fmaxf(sqrtf(red[0]), EPSF);
  for (int i = threadIdx.x; i < 4096; i += 256) {
    float v = xr[i] * inv;
    unsigned short h = f32_bf16_rne(v);
    float lo = v - bf16_f32(h);
    ah[(size_t)row * 4096 + i] = h;
    al[(size_t)row * 4096 + i] = f32_bf16_rne(lo);
  }
}

// ---------------- K2: column sum of squares of W ----------------
__global__ void k_colsum_w2(const float* __restrict__ w, float* __restrict__ s_w2) {
  const int j = blockIdx.x * 256 + threadIdx.x;
  const int o0 = blockIdx.y * 256;
  float s = 0.f;
  for (int r = 0; r < 256; ++r) {
    float v = w[(size_t)(o0 + r) * 4096 + j];
    s += v * v;
  }
  atomicAdd(&s_w2[j], s);
}

// ---------------- K3: build injected_pre^T (bf16 split) + column sumsq ----------------
// Bt[j][o] = W[o,j]/max(sw[j],eps)*(1-ql[o]) + Q[j,o]*ql[o]
__global__ void k_build_injected(const float* __restrict__ w,
                                 const float* __restrict__ q,
                                 const int* __restrict__ queue_iters,
                                 const int* __restrict__ iters,
                                 const float* __restrict__ s_w2,
                                 unsigned short* __restrict__ bh,
                                 unsigned short* __restrict__ bl,
                                 float* __restrict__ s_inj2) {
  __shared__ float wT[32][33];
  const int o0 = blockIdx.x * 32;
  const int j0 = blockIdx.y * 32;
  const int tx = threadIdx.x;  // 0..31 (one wave per ty row)
  const int ty = threadIdx.y;  // 0..7
  #pragma unroll
  for (int r = 0; r < 4; ++r) {
    int ol = ty + 8 * r;
    wT[tx][ol] = w[(size_t)(o0 + ol) * 4096 + (j0 + tx)];
  }
  __syncthreads();
  const int it = iters[0] + 1;
  const int o = o0 + tx;
  const float ql = ((it > 8000) && ((it - queue_iters[o]) <= 200)) ? 0.15f : 0.f;
  #pragma unroll
  for (int r = 0; r < 4; ++r) {
    int jl = ty + 8 * r;
    int j = j0 + jl;
    float invsw = 1.f / fmaxf(sqrtf(s_w2[j]), EPSF);
    float val = wT[jl][tx] * invsw * (1.f - ql) + q[(size_t)j * 4096 + o] * ql;
    float v2 = val * val;  // wave32 reduction over tx
    #pragma unroll
    for (int m = 16; m > 0; m >>= 1) v2 += __shfl_xor(v2, m, 32);
    if (tx == 0) atomicAdd(&s_inj2[j], v2);
    unsigned short h = f32_bf16_rne(val);
    float lo = val - bf16_f32(h);
    bh[(size_t)j * 4096 + o] = h;
    bl[(size_t)j * 4096 + o] = f32_bf16_rne(lo);
  }
}

// ---------------- K4: GEMM out[b,j] = sum_k A[b,k]*Bt[j,k], epilogue column scale ----
// Block: 256 threads (8 waves), tile 64(M) x 128(N), K-step 32.
// Double-buffered LDS, staged with GLOBAL_LOAD_ASYNC_TO_LDS_B128 (ASYNCcnt).
// 3x bf16-split products accumulated in f32 WMMA.
__global__ void __launch_bounds__(256)
k_gemm(const unsigned short* __restrict__ ah,
       const unsigned short* __restrict__ al,
       const unsigned short* __restrict__ bh,
       const unsigned short* __restrict__ bl,
       const float* __restrict__ s_inj2,
       float* __restrict__ out) {
  const int tid = threadIdx.x;
  const int lane = tid & 31;
  const int wave = tid >> 5;
  const int wm = wave & 1;
  const int wn = wave >> 1;
  const int m_block = blockIdx.y * 64;
  const int n_block = blockIdx.x * 128;

  v8f acc[2][2];
  #pragma unroll
  for (int i = 0; i < 2; ++i)
    #pragma unroll
    for (int j = 0; j < 2; ++j) acc[i][j] = v8f{};

  const int srow = tid >> 2;       // 0..63
  const int schk = (tid & 3) * 8;  // element offset 0,8,16,24 (16B chunks)
  const int am = lane & 15;
  const int a_koff = (lane < 16) ? 0 : 8;   // A frag: K {0..7,16..23} / {8..15,24..31}
  const int b_koff = (lane < 16) ? 0 : 8;   // B frag: K {0..7,16..23} / {8..15,24..31}

#if USE_ASYNC_LDS
  // padded stride 40 elements (80B) -> conflict-free b128 frag loads
  __shared__ __align__(16) unsigned short Ash[2][64 * 40];
  __shared__ __align__(16) unsigned short Asl[2][64 * 40];
  __shared__ __align__(16) unsigned short Bsh[2][128 * 40];
  __shared__ __align__(16) unsigned short Bsl[2][128 * 40];

  // per-thread global source pointers (advance by 32 elements per K-step)
  const unsigned short* gah = ah + (size_t)(m_block + srow) * 4096 + schk;
  const unsigned short* gal = al + (size_t)(m_block + srow) * 4096 + schk;
  const unsigned short* gb0h = bh + (size_t)(n_block + srow) * 4096 + schk;
  const unsigned short* gb1h = bh + (size_t)(n_block + srow + 64) * 4096 + schk;
  const unsigned short* gb0l = bl + (size_t)(n_block + srow) * 4096 + schk;
  const unsigned short* gb1l = bl + (size_t)(n_block + srow + 64) * 4096 + schk;

  unsigned lAh[2], lAl[2], lB0h[2], lB1h[2], lB0l[2], lB1l[2];
  #pragma unroll
  for (int b = 0; b < 2; ++b) {
    lAh[b]  = (unsigned)(uintptr_t)&Ash[b][srow * 40 + schk];
    lAl[b]  = (unsigned)(uintptr_t)&Asl[b][srow * 40 + schk];
    lB0h[b] = (unsigned)(uintptr_t)&Bsh[b][srow * 40 + schk];
    lB1h[b] = (unsigned)(uintptr_t)&Bsh[b][(srow + 64) * 40 + schk];
    lB0l[b] = (unsigned)(uintptr_t)&Bsl[b][srow * 40 + schk];
    lB1l[b] = (unsigned)(uintptr_t)&Bsl[b][(srow + 64) * 40 + schk];
  }

  // prologue: stage tile 0 into buffer 0
  async_cp16(gah, lAh[0]);
  async_cp16(gal, lAl[0]);
  async_cp16(gb0h, lB0h[0]);
  async_cp16(gb1h, lB1h[0]);
  async_cp16(gb0l, lB0l[0]);
  async_cp16(gb1l, lB1l[0]);

  #pragma unroll 2
  for (int it = 0; it < 128; ++it) {
    const int cur = it & 1;
    WAIT_ASYNC0();        // my tile-`it` loads have landed in LDS
    __syncthreads();      // everyone's have; everyone done reading buf[cur^1]
    if (it + 1 < 128) {   // stage tile it+1 into the other buffer
      const int koff = (it + 1) * 32;
      const int nxt = cur ^ 1;
      async_cp16(gah + koff, lAh[nxt]);
      async_cp16(gal + koff, lAl[nxt]);
      async_cp16(gb0h + koff, lB0h[nxt]);
      async_cp16(gb1h + koff, lB1h[nxt]);
      async_cp16(gb0l + koff, lB0l[nxt]);
      async_cp16(gb1l + koff, lB1l[nxt]);
    }
    // ---- fragment loads from buf[cur] ----
    FragBF afh[2], afl[2], bfh[2], bfl[2];
    #pragma unroll
    for (int mf = 0; mf < 2; ++mf) {
      const int row = wm * 32 + mf * 16 + am;
      const unsigned short* pa = &Ash[cur][row * 40 + a_koff];
      afh[mf].u[0] = *(const uint4*)pa;
      afh[mf].u[1] = *(const uint4*)(pa + 16);
      const unsigned short* pl = &Asl[cur][row * 40 + a_koff];
      afl[mf].u[0] = *(const uint4*)pl;
      afl[mf].u[1] = *(const uint4*)(pl + 16);
    }
    #pragma unroll
    for (int nf = 0; nf < 2; ++nf) {
      const int rowb = wn * 32 + nf * 16 + am;
      const unsigned short* pb = &Bsh[cur][rowb * 40 + b_koff];
      bfh[nf].u[0] = *(const uint4*)pb;
      bfh[nf].u[1] = *(const uint4*)(pb + 16);
      const unsigned short* pbl = &Bsl[cur][rowb * 40 + b_koff];
      bfl[nf].u[0] = *(const uint4*)pbl;
      bfl[nf].u[1] = *(const uint4*)(pbl + 16);
    }
    // ---- 3-term bf16-split WMMA accumulation ----
    #pragma unroll
    for (int mf = 0; mf < 2; ++mf)
      #pragma unroll
      for (int nf = 0; nf < 2; ++nf) {
        acc[mf][nf] = __builtin_amdgcn_wmma_f32_16x16x32_bf16(
            false, afh[mf].v, false, bfh[nf].v, (short)0, acc[mf][nf], false, false);
        acc[mf][nf] = __builtin_amdgcn_wmma_f32_16x16x32_bf16(
            false, afh[mf].v, false, bfl[nf].v, (short)0, acc[mf][nf], false, false);
        acc[mf][nf] = __builtin_amdgcn_wmma_f32_16x16x32_bf16(
            false, afl[mf].v, false, bfh[nf].v, (short)0, acc[mf][nf], false, false);
      }
  }
#else
  // -------- fallback: synchronous staging, single buffer --------
  __shared__ __align__(16) unsigned short Ash[64 * 40];
  __shared__ __align__(16) unsigned short Asl[64 * 40];
  __shared__ __align__(16) unsigned short Bsh[128 * 40];
  __shared__ __align__(16) unsigned short Bsl[128 * 40];

  for (int k0 = 0; k0 < 4096; k0 += 32) {
    *(uint4*)&Ash[srow * 40 + schk] =
        *(const uint4*)&ah[(size_t)(m_block + srow) * 4096 + k0 + schk];
    *(uint4*)&Asl[srow * 40 + schk] =
        *(const uint4*)&al[(size_t)(m_block + srow) * 4096 + k0 + schk];
    #pragma unroll
    for (int h = 0; h < 2; ++h) {
      int r = srow + h * 64;
      *(uint4*)&Bsh[r * 40 + schk] =
          *(const uint4*)&bh[(size_t)(n_block + r) * 4096 + k0 + schk];
      *(uint4*)&Bsl[r * 40 + schk] =
          *(const uint4*)&bl[(size_t)(n_block + r) * 4096 + k0 + schk];
    }
    __syncthreads();
    FragBF afh[2], afl[2], bfh[2], bfl[2];
    #pragma unroll
    for (int mf = 0; mf < 2; ++mf) {
      const int row = wm * 32 + mf * 16 + am;
      const unsigned short* pa = &Ash[row * 40 + a_koff];
      afh[mf].u[0] = *(const uint4*)pa;
      afh[mf].u[1] = *(const uint4*)(pa + 16);
      const unsigned short* pl = &Asl[row * 40 + a_koff];
      afl[mf].u[0] = *(const uint4*)pl;
      afl[mf].u[1] = *(const uint4*)(pl + 16);
    }
    #pragma unroll
    for (int nf = 0; nf < 2; ++nf) {
      const int rowb = wn * 32 + nf * 16 + am;
      const unsigned short* pb = &Bsh[rowb * 40 + b_koff];
      bfh[nf].u[0] = *(const uint4*)pb;
      bfh[nf].u[1] = *(const uint4*)(pb + 16);
      const unsigned short* pbl = &Bsl[rowb * 40 + b_koff];
      bfl[nf].u[0] = *(const uint4*)pbl;
      bfl[nf].u[1] = *(const uint4*)(pbl + 16);
    }
    #pragma unroll
    for (int mf = 0; mf < 2; ++mf)
      #pragma unroll
      for (int nf = 0; nf < 2; ++nf) {
        acc[mf][nf] = __builtin_amdgcn_wmma_f32_16x16x32_bf16(
            false, afh[mf].v, false, bfh[nf].v, (short)0, acc[mf][nf], false, false);
        acc[mf][nf] = __builtin_amdgcn_wmma_f32_16x16x32_bf16(
            false, afh[mf].v, false, bfl[nf].v, (short)0, acc[mf][nf], false, false);
        acc[mf][nf] = __builtin_amdgcn_wmma_f32_16x16x32_bf16(
            false, afl[mf].v, false, bfh[nf].v, (short)0, acc[mf][nf], false, false);
      }
    __syncthreads();
  }
#endif

  // ---- epilogue: divide each output column by max(||inj col||,eps) ----
  // C/D f32 frag: VGPR v -> lanes0-15 M=v, lanes16-31 M=v+8; N = lane&15
  #pragma unroll
  for (int nf = 0; nf < 2; ++nf) {
    const int col = n_block + wn * 32 + nf * 16 + (lane & 15);
    const float scale = 1.f / fmaxf(sqrtf(s_inj2[col]), EPSF);
    #pragma unroll
    for (int mf = 0; mf < 2; ++mf) {
      const int rbase = m_block + wm * 32 + mf * 16 + ((lane < 16) ? 0 : 8);
      #pragma unroll
      for (int v = 0; v < 8; ++v) {
        out[(size_t)(rbase + v) * 4096 + col] = acc[mf][nf][v] * scale;
      }
    }
  }
}

extern "C" void kernel_launch(void* const* d_in, const int* in_sizes, int n_in,
                              void* d_out, int out_size, void* d_ws, size_t ws_size,
                              hipStream_t stream) {
  const float* x  = (const float*)d_in[0];         // 512x4096
  const float* w  = (const float*)d_in[1];         // 4096x4096
  const float* q  = (const float*)d_in[2];         // 4096x4096
  const int* queue_iters = (const int*)d_in[3];    // 4096
  const int* iters = (const int*)d_in[4];          // 1
  float* out = (float*)d_out;                      // 512x4096

  char* ws = (char*)d_ws;
  float* s_w2   = (float*)(ws);
  float* s_inj2 = (float*)(ws + 16384);
  unsigned short* ah = (unsigned short*)(ws + 32768);
  unsigned short* al = ah + (size_t)512 * 4096;
  unsigned short* bh = al + (size_t)512 * 4096;
  unsigned short* bl = bh + (size_t)4096 * 4096;

  k_zero<<<32, 256, 0, stream>>>((float*)ws, 8192);
  k_norm_split_x<<<512, 256, 0, stream>>>(x, ah, al);
  k_colsum_w2<<<dim3(16, 16), 256, 0, stream>>>(w, s_w2);
  k_build_injected<<<dim3(128, 128), dim3(32, 8), 0, stream>>>(
      w, q, queue_iters, iters, s_w2, bh, bl, s_inj2);
  k_gemm<<<dim3(32, 8), 256, 0, stream>>>(ah, al, bh, bl, s_inj2, out);
}
